// CharRNN_13288628814384
// MI455X (gfx1250) — matrix operand order repaired and checked by reading, not compiled, not run on hardware
//
#include <hip/hip_runtime.h>
#include <stdint.h>

// ---------------------------------------------------------------------------
// CharRNN fused kernel for gfx1250 (MI455X), wave32 + V_WMMA_F32_16X16X32_BF16
//
//   x:[128,1024] int, emb:[128,16], W_ih:[512,16], W_hh:[512,512],
//   b_ih,b_hh:[512], W_fc:[128,512], b_fc:[128]  -> logits [128,1024,128] f32
//
// Strategy: batch rows are independent in the recurrence -> 8 persistent
// workgroups, each owning a 16-row slice, loop T=1024 steps in-kernel.
// Weights pre-packed (separate kernel) into bf16 tiles in the exact WMMA
// B-operand lane layout, streamed from L2 every step. A per-iteration memory
// clobber stops LICM from hoisting the (loop-invariant) tile loads into
// registers — that hoist spilled to scratch in the previous revision.
// h kept in LDS; A operands preloaded into registers per phase so the
// 68 recurrent WMMAs issue back-to-back.
// ---------------------------------------------------------------------------

typedef __attribute__((ext_vector_type(16))) __bf16 v16bf;
typedef __attribute__((ext_vector_type(8)))  float  v8f;

#define HIDN   512
#define EMBN   16
#define VOCABN 128
#define TLEN   1024
#define KTOT   17      // ceil((512+16)/32) K-tiles for [W_hh | W_ih | 0]
#define NT_H   32      // HID/16 N-tiles
#define NT_V   8       // VOCAB/16 N-tiles
#define ROWS   546     // LDS h row length in bf16 elems (544 used + pad), even
#define ROWD   273     // row length in dwords (odd -> bank-conflict-free A loads)

__device__ __forceinline__ unsigned short f2bf(float f) {
  union { __bf16 b; unsigned short s; } c;
  c.b = (__bf16)f;                                  // native cvt, RNE
  return c.s;
}

__device__ __forceinline__ float fast_tanh(float x) {
  // tanh(x) = 1 - 2/(exp(2x)+1); saturates correctly at +-1
  return 1.0f - __fdividef(2.0f, __expf(2.0f * x) + 1.0f);
}

// Load 16x32 bf16 A-tile (K window [32*kt, 32*kt+32)) from LDS h matrix
// (row-major [16][ROWS] bf16) into WMMA A-operand lanes in contiguous K order (same as packed B):
//   lane<16:  M=lane, K = {2v,2v+1} (v<8)
//   lane>=16: same M, K offset by +16.
__device__ __forceinline__ v16bf load_A(const uint32_t* hl, int kt, int lane) {
  const int m    = lane & 15;
  const int kb2  = (lane >> 4) * 8;                 // dword offset of K-base
  const int base = m * ROWD + kt * 16 + kb2;
  union { v16bf v; uint32_t u[8]; } A;
#pragma unroll
  for (int v = 0; v < 8; ++v)
    A.u[v] = hl[base + v];
  return A.v;
}

// B tiles are pre-packed: 1 KB per tile = 32 lanes x 32 bytes, lane-major.
__device__ __forceinline__ v16bf load_B(const uint4* tile, int lane) {
  union { v16bf v; uint4 q[2]; } B;
  B.q[0] = tile[lane * 2 + 0];
  B.q[1] = tile[lane * 2 + 1];
  return B.v;
}

// ---------------------------------------------------------------------------
// Pre-pass: pack [W_hh | W_ih | 0] and W_fc into bf16 WMMA B-operand tiles.
// B packing, contiguous K order matching load_A (32x16 bf16): lane n = lane&15 (N col), K base =
// (lane>>4)*16, dword v holds K = base+2v (lo16) and base+2v+1 (hi16).
// Tile index: hidden  tile = nt*17 + kt  (nt<32, kt<17)
//             logits  tile = nt*16 + kt  (nt<8,  kt<16)
// ---------------------------------------------------------------------------
__global__ __launch_bounds__(256) void pack_weights(
    const float* __restrict__ W_ih, const float* __restrict__ W_hh,
    const float* __restrict__ W_fc,
    uint32_t* __restrict__ wsW32, uint32_t* __restrict__ wsF32) {
  const int gid  = blockIdx.x * blockDim.x + threadIdx.x;
  const int lane = gid & 31;
  const int tile = gid >> 5;
  if (tile >= NT_H * KTOT + NT_V * 16) return;

  if (tile < NT_H * KTOT) {                       // hidden weights
    const int nt = tile / KTOT, kt = tile % KTOT;
    const int n  = nt * 16 + (lane & 15);
    const int kb = kt * 32 + (lane >> 4) * 16;
#pragma unroll
    for (int v = 0; v < 8; ++v) {
      unsigned short lo, hi;
      {
        const int k = kb + 2 * v;
        float f = (k < HIDN) ? W_hh[n * HIDN + k]
                : (k < HIDN + EMBN) ? W_ih[n * EMBN + (k - HIDN)] : 0.f;
        lo = f2bf(f);
      }
      {
        const int k = kb + 2 * v + 1;
        float f = (k < HIDN) ? W_hh[n * HIDN + k]
                : (k < HIDN + EMBN) ? W_ih[n * EMBN + (k - HIDN)] : 0.f;
        hi = f2bf(f);
      }
      wsW32[tile * 256 + lane * 8 + v] = (uint32_t)lo | ((uint32_t)hi << 16);
    }
  } else {                                        // classifier weights
    const int t2 = tile - NT_H * KTOT;
    const int nt = t2 / 16, kt = t2 % 16;
    const int n  = nt * 16 + (lane & 15);         // < 128
    const int kb = kt * 32 + (lane >> 4) * 16;
#pragma unroll
    for (int v = 0; v < 8; ++v) {
      const int k = kb + 2 * v;
      unsigned short lo = f2bf(W_fc[n * HIDN + k]);
      unsigned short hi = f2bf(W_fc[n * HIDN + k + 1]);
      wsF32[t2 * 256 + lane * 8 + v] = (uint32_t)lo | ((uint32_t)hi << 16);
    }
  }
}

// ---------------------------------------------------------------------------
// Main fused RNN: 8 workgroups x 256 threads (8 wave32). WG b owns batch rows
// [16b, 16b+16). Wave w accumulates hidden N-tiles {4w..4w+3} and logits
// N-tile w. h (+ concatenated emb columns 512..527, zeros 528..543) in LDS.
// ---------------------------------------------------------------------------
__global__ __launch_bounds__(256) void rnn_fused(
    const int*   __restrict__ x,    const float* __restrict__ emb,
    const float* __restrict__ b_ih, const float* __restrict__ b_hh,
    const float* __restrict__ b_fc,
    const uint32_t* __restrict__ wsW32, const uint32_t* __restrict__ wsF32,
    float* __restrict__ out) {
  __shared__ uint32_t hl[16 * ROWD];              // 17472 B hidden state
  __shared__ float    emb_lds[VOCABN * EMBN];     // 8 KB embedding table
  uint16_t* h16 = (uint16_t*)hl;

  const int tid  = threadIdx.x;
  const int lane = tid & 31;
  const int wv   = tid >> 5;                      // wave id 0..7
  const int b0   = blockIdx.x * 16;

  // zero h0 / emb cols / padding, stage embedding table in LDS
  for (int i = tid; i < 16 * ROWD; i += 256) hl[i] = 0;
  for (int i = tid; i < VOCABN * EMBN; i += 256) emb_lds[i] = emb[i];

  // hoist biases (per-lane N column is fixed for the whole scan)
  float biasR[4];
#pragma unroll
  for (int j = 0; j < 4; ++j)
    biasR[j] = b_ih[(wv * 4 + j) * 16 + (lane & 15)]
             + b_hh[(wv * 4 + j) * 16 + (lane & 15)];
  const float biasF = b_fc[wv * 16 + (lane & 15)];

  const uint4* wsW4 = (const uint4*)wsW32;
  const uint4* wsF4 = (const uint4*)wsF32;

  const int m_e = tid >> 4;                       // row this thread embeds
  const int e_e = tid & 15;                       // emb element
  const int xbase = (b0 + m_e) * TLEN;
  const int mhi = (lane >> 4) * 8;                // C/D row base for this lane
  __syncthreads();

  for (int t = 0; t < TLEN; ++t) {
    // Defeat cross-iteration LICM: without this, all (loop-invariant) weight
    // tile loads get hoisted out of the scan, overflow the register file and
    // spill to scratch. With it, tiles stream from L2 each step and the
    // in-iteration scheduler still prefetches them ahead of the WMMAs.
    asm volatile("" ::: "memory");

    // ---- phase 1: scatter emb row of step t into h columns 512..527 ----
    {
      const int idx = x[xbase + t];
      h16[m_e * ROWS + HIDN + e_e] = f2bf(emb_lds[idx * EMBN + e_e]);
    }
    __syncthreads();

    // ---- phase 2: h_new = tanh([h|e]·[W_hh|W_ih]^T + b) ----
    v8f acc[4];
#pragma unroll
    for (int j = 0; j < 4; ++j)
#pragma unroll
      for (int v = 0; v < 8; ++v) acc[j][v] = biasR[j];

    // preload ALL A tiles: DS loads pipeline under one wait, WMMAs then
    // issue back-to-back against the streamed L2 B tiles.
    v16bf areg[KTOT];
#pragma unroll
    for (int kt = 0; kt < KTOT; ++kt) areg[kt] = load_A(hl, kt, lane);

#pragma unroll
    for (int kt = 0; kt < KTOT; ++kt) {
#pragma unroll
      for (int j = 0; j < 4; ++j) {
        const int nt = wv * 4 + j;
        const v16bf b = load_B(wsW4 + (size_t)(nt * KTOT + kt) * 64, lane);
        acc[j] = __builtin_amdgcn_wmma_f32_16x16x32_bf16(
            false, areg[kt], false, b, (short)0, acc[j], false, false);
      }
    }
#pragma unroll
    for (int j = 0; j < 4; ++j)
#pragma unroll
      for (int v = 0; v < 8; ++v) acc[j][v] = fast_tanh(acc[j][v]);

    __syncthreads();                              // old-h reads complete
#pragma unroll
    for (int j = 0; j < 4; ++j) {
      const int n = (wv * 4 + j) * 16 + (lane & 15);
#pragma unroll
      for (int v = 0; v < 8; ++v)
        h16[(mhi + v) * ROWS + n] = f2bf(acc[j][v]);
    }
    __syncthreads();                              // h_new visible in LDS

    // ---- phase 3: logits = h_new·W_fc^T + b_fc (wave w -> N-tile w) ----
    v8f c2;
#pragma unroll
    for (int v = 0; v < 8; ++v) c2[v] = biasF;

    v16bf areg2[16];
#pragma unroll
    for (int kt = 0; kt < 16; ++kt) areg2[kt] = load_A(hl, kt, lane);

#pragma unroll
    for (int kt = 0; kt < 16; ++kt) {
      const v16bf b = load_B(wsF4 + (size_t)(wv * 16 + kt) * 64, lane);
      c2 = __builtin_amdgcn_wmma_f32_16x16x32_bf16(
          false, areg2[kt], false, b, (short)0, c2, false, false);
    }
    {
      const int n = wv * 16 + (lane & 15);
#pragma unroll
      for (int v = 0; v < 8; ++v)
        out[((size_t)(b0 + mhi + v) * TLEN + t) * VOCABN + n] = c2[v];
    }
    __syncthreads();                              // logits reads done before next step
  }
}

// ---------------------------------------------------------------------------
extern "C" void kernel_launch(void* const* d_in, const int* in_sizes, int n_in,
                              void* d_out, int out_size, void* d_ws, size_t ws_size,
                              hipStream_t stream) {
  const int*   x    = (const int*)  d_in[0];
  const float* emb  = (const float*)d_in[1];
  const float* W_ih = (const float*)d_in[2];
  const float* W_hh = (const float*)d_in[3];
  const float* b_ih = (const float*)d_in[4];
  const float* b_hh = (const float*)d_in[5];
  const float* W_fc = (const float*)d_in[6];
  const float* b_fc = (const float*)d_in[7];
  float* out = (float*)d_out;

  // workspace: 544 hidden tiles + 128 fc tiles, 1 KB each (672 KB total)
  uint32_t* wsW32 = (uint32_t*)d_ws;
  uint32_t* wsF32 = wsW32 + (size_t)NT_H * KTOT * 256;

  // 672 tiles * 32 lanes = 21504 threads
  pack_weights<<<84, 256, 0, stream>>>(W_ih, W_hh, W_fc, wsW32, wsF32);
  rnn_fused<<<8, 256, 0, stream>>>(x, emb, b_ih, b_hh, b_fc, wsW32, wsF32, out);
}
